// LinAngularAttention_binary_85676007621266
// MI455X (gfx1250) — matrix operand order repaired and checked by reading, not compiled, not run on hardware
//
#include <hip/hip_runtime.h>
#include <hip/hip_bf16.h>

typedef __attribute__((ext_vector_type(16))) _Float16 v16h;
typedef __attribute__((ext_vector_type(8)))  _Float16 v8h;
typedef __attribute__((ext_vector_type(8)))  float    v8f;
typedef int v4i_g __attribute__((vector_size(16)));   // matches builtin prototype pointee

#define CAT16(lo, hi) __builtin_shufflevector(lo, hi, 0,1,2,3,4,5,6,7,8,9,10,11,12,13,14,15)

#if __has_builtin(__builtin_amdgcn_global_load_async_to_lds_b128) && \
    __has_builtin(__builtin_amdgcn_s_wait_asynccnt)
#define HAVE_ASYNC_LDS 1
typedef __attribute__((address_space(1))) v4i_g* glob_v4i_ptr;
typedef __attribute__((address_space(3))) v4i_g* lds_v4i_ptr;
#else
#define HAVE_ASYNC_LDS 0
#endif

static constexpr int BQ    = 16;                 // batch
static constexpr int NTOK  = 4096;               // tokens
static constexpr int MROWS = BQ * NTOK;          // 65536
static constexpr size_t QCNT = (size_t)MROWS * 256;  // 16777216 q (and k) elements
static constexpr int LDST  = 40;                 // LDS row stride (halves): 80B, 16B-aligned, bank-spread

// ---------------------------------------------------------------------------
// Tiled WMMA GEMM, double-buffered pipeline: C[M,N] = A[M,256] * W[N,256]^T (+bias)
// A32: A is fp32 (converted to f16 while staging).  OUT16: store f16 else f32.
// 256 threads (8 waves), block tile 128x128, BK=32, wave tile 64x32.
// ---------------------------------------------------------------------------
template<bool A32, bool OUT16>
__global__ __launch_bounds__(256)
void gemm_f16_wmma(const void* __restrict__ Aptr, const float* __restrict__ W,
                   void* __restrict__ Cptr, const float* __restrict__ bias,
                   int lda, int ldc)
{
    __shared__ __attribute__((aligned(16))) _Float16 sA[2][128 * LDST];
    __shared__ __attribute__((aligned(16))) _Float16 sB[2][128 * LDST];

    const int t    = threadIdx.x;
    const int lane = t & 31;
    const int wv   = t >> 5;
    const int wm   = wv & 1;          // 64-row half
    const int wn   = wv >> 1;         // 32-col quarter
    const int bm   = blockIdx.y * 128;
    const int bn   = blockIdx.x * 128;

    const int ar = t >> 1;            // staging row 0..127
    const int ak = (t & 1) * 16;      // staging 16-half segment

    v8h a0, a1, b0, b1;               // register staging (tile in flight)

    auto fetch = [&](int k0) {
        if (A32) {
            const float* src = (const float*)Aptr + (size_t)(bm + ar) * lda + k0 + ak;
            for (int i = 0; i < 8; ++i) { a0[i] = (_Float16)src[i]; a1[i] = (_Float16)src[8 + i]; }
        } else {
            const _Float16* src = (const _Float16*)Aptr + (size_t)(bm + ar) * lda + k0 + ak;
            a0 = *(const v8h*)src; a1 = *(const v8h*)(src + 8);
        }
        const float* wsrc = W + (size_t)(bn + ar) * 256 + k0 + ak;
        for (int i = 0; i < 8; ++i) { b0[i] = (_Float16)wsrc[i]; b1[i] = (_Float16)wsrc[8 + i]; }
    };
    auto stage = [&](int buf) {
        *(v8h*)(sA[buf] + ar * LDST + ak)     = a0;
        *(v8h*)(sA[buf] + ar * LDST + ak + 8) = a1;
        *(v8h*)(sB[buf] + ar * LDST + ak)     = b0;
        *(v8h*)(sB[buf] + ar * LDST + ak + 8) = b1;
    };

    fetch(0);
    stage(0);
    fetch(32);
    __syncthreads();

    v8f acc[4][2] = {};
    const int abase = (lane >> 4) * 8;      // A: K chunk {0|8}, then +16
    const int bbase = (lane >> 4) * 8;      // B: K chunk {0|8}, then +16

    for (int ks = 0; ks < 8; ++ks) {
        const int buf = ks & 1;
        if (ks + 1 < 8) stage(buf ^ 1);           // LDS write of tile ks+1 (other buffer)
        if (ks + 2 < 8) fetch((ks + 2) * 32);     // global fetch of tile ks+2 overlaps WMMA

        v16h bfrag[2];
        for (int ni = 0; ni < 2; ++ni) {
            const _Float16* p = sB[buf] + (wn * 32 + ni * 16 + (lane & 15)) * LDST + bbase;
            v8h lo = *(const v8h*)p;
            v8h hi = *(const v8h*)(p + 16);
            bfrag[ni] = CAT16(lo, hi);
        }
        for (int mi = 0; mi < 4; ++mi) {
            const _Float16* p = sA[buf] + (wm * 64 + mi * 16 + (lane & 15)) * LDST;
            v8h lo = *(const v8h*)(p + abase);
            v8h hi = *(const v8h*)(p + abase + 16);
            v16h afrag = CAT16(lo, hi);
            for (int ni = 0; ni < 2; ++ni)
                acc[mi][ni] = __builtin_amdgcn_wmma_f32_16x16x32_f16(
                    false, afrag, false, bfrag[ni], (short)0, acc[mi][ni], false, false);
        }
        __syncthreads();
    }

    // C/D layout: row = base + 8*(lane>=16) + vgpr, col = lane&15
    for (int mi = 0; mi < 4; ++mi)
        for (int ni = 0; ni < 2; ++ni) {
            const int row = bm + wm * 64 + mi * 16 + (lane >> 4) * 8;
            const int col = bn + wn * 32 + ni * 16 + (lane & 15);
            const float bb = bias ? bias[col] : 0.0f;
            for (int r = 0; r < 8; ++r) {
                float v = acc[mi][ni][r] + bb;
                if (OUT16) ((_Float16*)Cptr)[(size_t)(row + r) * ldc + col] = (_Float16)v;
                else       ((float*)Cptr)[(size_t)(row + r) * ldc + col]   = v;
            }
        }
}

// ---------------------------------------------------------------------------
// Global binarization means: sum |q_hat|, |k_hat| (per-head L2 normalized).
// One wave (32 lanes) == one head group of 32 channels (wave32).
// ---------------------------------------------------------------------------
__global__ __launch_bounds__(256)
void sum_abs_kernel(const _Float16* __restrict__ qkv, float* __restrict__ sums)
{
    __shared__ float red[16];
    float aq = 0.f, ak = 0.f;
    const size_t stride = (size_t)gridDim.x * 256;
    for (size_t idx = (size_t)blockIdx.x * 256 + threadIdx.x; idx < QCNT; idx += stride) {
        const size_t row = idx >> 8;
        const int    c   = (int)(idx & 255);
        if (idx + stride < QCNT) {
            __builtin_prefetch(qkv + ((idx + stride) >> 8) * 768 + ((idx + stride) & 255), 0, 1);
        }
        float q = (float)qkv[row * 768 + c];
        float k = (float)qkv[row * 768 + 256 + c];
        float sq = q * q, sk = k * k;
        for (int m = 1; m < 32; m <<= 1) { sq += __shfl_xor(sq, m, 32); sk += __shfl_xor(sk, m, 32); }
        aq += fabsf(q) * rsqrtf(sq);
        ak += fabsf(k) * rsqrtf(sk);
    }
    for (int m = 1; m < 32; m <<= 1) { aq += __shfl_xor(aq, m, 32); ak += __shfl_xor(ak, m, 32); }
    const int w = threadIdx.x >> 5, lane = threadIdx.x & 31;
    if (lane == 0) { red[w] = aq; red[8 + w] = ak; }
    __syncthreads();
    if (threadIdx.x == 0) {
        float s0 = 0.f, s1 = 0.f;
        for (int i = 0; i < 8; ++i) { s0 += red[i]; s1 += red[8 + i]; }
        atomicAdd(&sums[0], s0);
        atomicAdd(&sums[1], s1);
    }
}

// ---------------------------------------------------------------------------
// Binarized k^T in f16: kT[bh][d][n] = (k[b,n,h,d] > 0) ? 1 : 0
// ---------------------------------------------------------------------------
__global__ __launch_bounds__(256)
void kbin_transpose_kernel(const _Float16* __restrict__ qkv, _Float16* __restrict__ kT)
{
    __shared__ _Float16 tile[32][33];
    const int bh = blockIdx.x;         // b*8+h
    const int b  = bh >> 3, h = bh & 7;
    const int n0 = blockIdx.y * 32;
    const int t  = threadIdx.x;
    const int d  = t & 31;
    const int nl = t >> 5;
    for (int p = 0; p < 4; ++p) {
        const int n = nl + p * 8;
        float kv = (float)qkv[(size_t)(b * NTOK + n0 + n) * 768 + 256 + h * 32 + d];
        tile[d][n] = (kv > 0.f) ? (_Float16)1.0f : (_Float16)0.0f;
    }
    __syncthreads();
    const int rl = t >> 5;
    for (int p = 0; p < 4; ++p) {
        const int row = rl + p * 8;
        kT[(size_t)(bh * 32 + row) * NTOK + n0 + d] = tile[row][d];
    }
}

// ---------------------------------------------------------------------------
// attn0[bh][d][e] = sum_n kT[bh][d][n] * v[b,n,h,e]   (split-K WMMA + atomics)
// 128 threads (4 waves); double-buffered; A-tile staged with gfx1250
// global_load_async_to_lds_b128 (ASYNCcnt) when the builtin is available.
// ---------------------------------------------------------------------------
__global__ __launch_bounds__(128)
void attn_kernel(const _Float16* __restrict__ kT, const _Float16* __restrict__ qkv,
                 float* __restrict__ attn0)
{
    __shared__ __attribute__((aligned(16))) _Float16 sA[2][32 * LDST]; // [d][n]
    __shared__ __attribute__((aligned(16))) _Float16 sB[2][32 * LDST]; // [e][n] (transposed v)
    const int t    = threadIdx.x;
    const int lane = t & 31;
    const int wv   = t >> 5;
    const int bh   = blockIdx.x;
    const int b    = bh >> 3, h = bh & 7;
    const int n0   = blockIdx.y * 512;
    const int m16  = wv & 1, n16 = wv >> 1;
    const int srow = t >> 2;           // 0..31
    const int sseg = (t & 3) * 8;      // 8-half segment

    v8h vreg;
    auto stageA = [&](int buf, int kk) {
        const _Float16* gsrc = kT + (size_t)(bh * 32 + srow) * NTOK + n0 + kk + sseg;
        _Float16* ldst = sA[buf] + srow * LDST + sseg;
#if HAVE_ASYNC_LDS
        __builtin_amdgcn_global_load_async_to_lds_b128(
            (glob_v4i_ptr)gsrc, (lds_v4i_ptr)ldst, 0, 0);
#else
        *(v8h*)ldst = *(const v8h*)gsrc;
#endif
    };
    auto fetchB = [&](int kk) {
        vreg = *(const v8h*)(qkv + (size_t)(b * NTOK + n0 + kk + srow) * 768 + 512 + h * 32 + sseg);
    };
    auto stageB = [&](int buf) {   // transpose v [n][e] -> LDS [e][n]
        for (int i = 0; i < 8; ++i) sB[buf][(sseg + i) * LDST + srow] = vreg[i];
    };

    stageA(0, 0);
    fetchB(0);
    stageB(0);
    fetchB(32);
#if HAVE_ASYNC_LDS
    __builtin_amdgcn_s_wait_asynccnt(0);
#endif
    __syncthreads();

    v8f acc = {};
    const int abase = (lane >> 4) * 8;
    const int bbase = (lane >> 4) * 8;

    for (int ks = 0; ks < 16; ++ks) {
        const int buf = ks & 1;
        const int kk  = ks * 32;
        if (ks + 1 < 16) { stageA(buf ^ 1, kk + 32); stageB(buf ^ 1); }
        if (ks + 2 < 16) fetchB(kk + 64);

        const _Float16* pa = sA[buf] + (m16 * 16 + (lane & 15)) * LDST;
        v8h alo = *(const v8h*)(pa + abase);
        v8h ahi = *(const v8h*)(pa + abase + 16);
        v16h af = CAT16(alo, ahi);
        const _Float16* pb = sB[buf] + (n16 * 16 + (lane & 15)) * LDST + bbase;
        v8h blo = *(const v8h*)pb;
        v8h bhi = *(const v8h*)(pb + 16);
        v16h bf = CAT16(blo, bhi);
        acc = __builtin_amdgcn_wmma_f32_16x16x32_f16(false, af, false, bf, (short)0, acc, false, false);

#if HAVE_ASYNC_LDS
        __builtin_amdgcn_s_wait_asynccnt(0);
#endif
        __syncthreads();
    }
    const int row = m16 * 16 + (lane >> 4) * 8;
    const int col = n16 * 16 + (lane & 15);
    float* dst = attn0 + (size_t)bh * 1024;
    for (int r = 0; r < 8; ++r) atomicAdd(dst + (row + r) * 32 + col, acc[r]);
}

// ---------------------------------------------------------------------------
// Per-row epilogue: out = L2norm(0.5 v + (mq*mk/pi) * bq @ attn0) + dconv(v)
// 256 threads = 8 waves; wave w == head w (32 lanes == head dim, wave32).
// ---------------------------------------------------------------------------
__global__ __launch_bounds__(256)
void rowout_kernel(const _Float16* __restrict__ qkv, const float* __restrict__ attn0,
                   const float* __restrict__ sums, const float* __restrict__ Wd,
                   _Float16* __restrict__ outp)
{
    __shared__ float bq[256];
    const int row = blockIdx.x;
    const int b = row >> 12, n = row & 4095;
    const int t = threadIdx.x;
    const int h = t >> 5, d = t & 31;

    float qv = (float)qkv[(size_t)row * 768 + t];
    float vv = (float)qkv[(size_t)row * 768 + 512 + t];
    bq[t] = (qv > 0.f) ? 1.f : 0.f;
    __syncthreads();

    const float* at = attn0 + (size_t)(b * 8 + h) * 1024;
    float qa = 0.f;
    for (int e = 0; e < 32; ++e) qa += bq[h * 32 + e] * at[e * 32 + d];

    const float inv = 1.0f / 16777216.0f;
    const float mq = sums[0] * inv, mk = sums[1] * inv;
    float o = 0.5f * vv + (0.318309886183790672f * mq * mk) * qa;

    float ss = o * o;
    for (int m = 1; m < 32; m <<= 1) ss += __shfl_xor(ss, m, 32);
    o *= rsqrtf(ss);

    for (int j = 0; j < 9; ++j) {
        const int nn = n + j - 4;
        if (nn >= 0 && nn < NTOK)
            o += Wd[h * 9 + j] * (float)qkv[(size_t)(b * NTOK + nn) * 768 + 512 + t];
    }
    outp[(size_t)row * 256 + t] = (_Float16)o;
}

__global__ void zero_kernel(float* p, int n)
{
    int i = blockIdx.x * 256 + threadIdx.x;
    if (i < n) p[i] = 0.f;
}

extern "C" void kernel_launch(void* const* d_in, const int* in_sizes, int n_in,
                              void* d_out, int out_size, void* d_ws, size_t ws_size,
                              hipStream_t stream)
{
    (void)in_sizes; (void)n_in; (void)out_size; (void)ws_size;
    const float* x     = (const float*)d_in[0];
    const float* Wq    = (const float*)d_in[1];
    const float* Wkv   = (const float*)d_in[2];
    const float* Wproj = (const float*)d_in[3];
    const float* bproj = (const float*)d_in[4];
    const float* Wd    = (const float*)d_in[5];
    float* out = (float*)d_out;

    char* ws = (char*)d_ws;
    _Float16* qkv  = (_Float16*)ws;                                     // 65536*768 f16 = 96 MB
    _Float16* kT   = (_Float16*)(ws + 100663296);                       // 128*32*4096 f16 = 32 MB
    _Float16* outp = (_Float16*)(ws + 100663296 + 33554432);            // 65536*256 f16 = 32 MB
    float*    attn0= (float*)   (ws + 100663296 + 33554432 + 33554432); // 128*1024 f32
    float*    sums = attn0 + 131072;                                    // 2 floats

    zero_kernel<<<(131074 + 255) / 256, 256, 0, stream>>>(attn0, 131074);

    // q = x @ Wq^T  -> qkv[:, 0:256)
    gemm_f16_wmma<true, true><<<dim3(2, MROWS / 128), 256, 0, stream>>>(
        x, Wq, qkv, nullptr, 256, 768);
    // [k|v] = x @ Wkv^T -> qkv[:, 256:768)
    gemm_f16_wmma<true, true><<<dim3(4, MROWS / 128), 256, 0, stream>>>(
        x, Wkv, qkv + 256, nullptr, 256, 768);

    sum_abs_kernel<<<2048, 256, 0, stream>>>(qkv, sums);
    kbin_transpose_kernel<<<dim3(128, NTOK / 32), 256, 0, stream>>>(qkv, kT);
    attn_kernel<<<dim3(128, 8), 128, 0, stream>>>(kT, qkv, attn0);
    rowout_kernel<<<MROWS, 256, 0, stream>>>(qkv, attn0, sums, Wd, outp);

    // final: out = outp @ Wproj^T + bproj   (fp32 out)
    gemm_f16_wmma<false, false><<<dim3(2, MROWS / 128), 256, 0, stream>>>(
        outp, Wproj, out, bproj, 256, 256);
}
